// DiversityDensity_53833120088165
// MI455X (gfx1250) — compile-verified, matrix operand order checked by reading, not run on hardware
//
#include <hip/hip_runtime.h>

// DiversityDensity on MI455X (gfx1250, wave32, WMMA).
//
// Core: fused GEMM(4096x65536, K=32) + per-row min, using split-f16
// (hi+lo) V_WMMA_F32_16X16X32_F16 (3 chained WMMAs ~= f32 accuracy).
// L is pre-scaled by -2 and ||l||^2 is injected as the WMMA C-initializer,
// so each chain's D output is directly  l_sq - 2*u.l.  Two L tiles per
// iteration are folded into the running min with one v_min3_num_f32
// (inline asm: avoids the minnum canonicalize bloat), and the next pair
// of tiles is preloaded at the top of the body (software pipelining).
// Data (8.5MB) lives in L2 (192MB); kernel is WMMA-throughput bound.

typedef __attribute__((ext_vector_type(16))) _Float16 v16h;
typedef __attribute__((ext_vector_type(8)))  float    v8f;

#define N_U 4096
#define N_L 65536
#define NZ  32
#define UT  (N_U / 16)   // 256 U tiles
#define LTI (N_L / 16)   // 4096 L tiles
#define MT  4            // U tiles kept in registers per block (64 rows)
#define LOG_2PI 1.8378770664093453f

// Monotone float <-> uint mapping: order-preserving for ALL floats,
// so atomicMin on uint == float min (deterministic under any order).
__device__ __forceinline__ unsigned mapf(float f) {
  unsigned b = __float_as_uint(f);
  return (b & 0x80000000u) ? ~b : (b | 0x80000000u);
}
__device__ __forceinline__ float unmapf(unsigned u) {
  unsigned b = (u & 0x80000000u) ? (u & 0x7FFFFFFFu) : ~u;
  return __uint_as_float(b);
}

// Single-instruction 3-way float min (no canonicalize ops emitted).
__device__ __forceinline__ float fmin3_raw(float a, float b, float c) {
  float d;
  asm("v_min3_num_f32 %0, %1, %2, %3" : "=v"(d) : "v"(a), "v"(b), "v"(c));
  return d;
}

__global__ void k_init(unsigned* __restrict__ rowmin, unsigned* __restrict__ mm) {
  int i = blockIdx.x * blockDim.x + threadIdx.x;
  if (i < N_U) rowmin[i] = 0xFFFFFFFFu;        // mapped +inf sentinel
  if (i == 0) { mm[0] = 0xFFFFFFFFu; mm[1] = 0u; }  // dd>=0 -> raw bits ordered
}

// Pack rows of Z (ntiles*16 x 32 f32) into CDNA5 16-bit A/B fragment layout:
//   lane m    (0..15): K = 0..7, 16..23  of row (16t+m)
//   lane m+16        : K = 8..15, 24..31 of row (16t+m)
// stored as one contiguous v16h (32B) per (tile,lane); value packed is
// scale*x split into f16 hi + f16 lo (scale*x = hi + lo to ~2^-22 rel).
// Row sum-of-squares uses the UNscaled x.
__global__ __launch_bounds__(256) void k_pack(
    const float* __restrict__ Z, v16h* __restrict__ Hi, v16h* __restrict__ Lo,
    float* __restrict__ sq, int ntiles, float scale)
{
  const int lane = threadIdx.x & 31;
  const int wave = threadIdx.x >> 5;
  const int t = blockIdx.x * 8 + wave;
  if (t >= ntiles) return;
  const int row = t * 16 + (lane & 15);
  const int kb  = (lane >> 4) * 8;
  const float* src = Z + (size_t)row * NZ;

  float x[16];
#pragma unroll
  for (int i = 0; i < 8; ++i) { x[i] = src[kb + i]; x[8 + i] = src[kb + 16 + i]; }

  v16h hi, lo;
  float s = 0.0f;
#pragma unroll
  for (int i = 0; i < 16; ++i) {
    float xv = x[i];
    s += xv * xv;
    float xs = xv * scale;               // exact for scale = 1 or -2
    _Float16 h = (_Float16)xs;
    hi[i] = h;
    lo[i] = (_Float16)(xs - (float)h);
  }
  Hi[(size_t)t * 32 + lane] = hi;
  Lo[(size_t)t * 32 + lane] = lo;
  s += __shfl_xor(s, 16, 32);                 // halves cover disjoint K
  if (lane < 16) sq[row] = s;
}

// Main: each block owns 64 U rows (4 A tiles, hi+lo in VGPRs) and one of 8
// N-splits. Each wave streams its 64 L tiles two at a time (32 iterations),
// preloading the next pair while 24 WMMAs retire; both tiles' chain outputs
// fold into the running min with a single v_min3_num_f32 per element.
__global__ __launch_bounds__(256) void k_wmma_min(
    const v16h* __restrict__ Uhi, const v16h* __restrict__ Ulo,
    const v16h* __restrict__ Lhi, const v16h* __restrict__ Llo,
    const float* __restrict__ lsq, unsigned* __restrict__ rowmin)
{
  const int lane = threadIdx.x & 31;
  const int wave = threadIdx.x >> 5;     // 0..7
  const int mblk = blockIdx.x;           // 0..63
  const int nspl = blockIdx.y;           // 0..7
  const int base = nspl * 8 + wave;      // first L tile for this wave

  v16h ahi[MT], alo[MT];
#pragma unroll
  for (int a = 0; a < MT; ++a) {
    int t = mblk * MT + a;
    ahi[a] = Uhi[(size_t)t * 32 + lane];
    alo[a] = Ulo[(size_t)t * 32 + lane];
  }

  float acc[MT][8];
#pragma unroll
  for (int a = 0; a < MT; ++a)
#pragma unroll
    for (int r = 0; r < 8; ++r) acc[a][r] = __builtin_inff();

  // Preload first pair of L tiles (fragments hold -2*L).
  int t0 = base;
  v16h b0h = Lhi[(size_t)t0 * 32 + lane], b0l = Llo[(size_t)t0 * 32 + lane];
  float ls0 = lsq[t0 * 16 + (lane & 15)];
  int t1 = t0 + 64;
  v16h b1h = Lhi[(size_t)t1 * 32 + lane], b1l = Llo[(size_t)t1 * 32 + lane];
  float ls1 = lsq[t1 * 16 + (lane & 15)];

  for (int it = 0; it < 32; ++it) {
    // Preload next pair (wave-uniform clamp on last iter: harmless re-read).
    int tn0 = t0 + 128; if (tn0 >= LTI) tn0 = base;
    int tn1 = tn0 + 64; if (tn1 >= LTI) tn1 = base;
    v16h n0h = Lhi[(size_t)tn0 * 32 + lane], n0l = Llo[(size_t)tn0 * 32 + lane];
    float nls0 = lsq[tn0 * 16 + (lane & 15)];
    v16h n1h = Lhi[(size_t)tn1 * 32 + lane], n1l = Llo[(size_t)tn1 * 32 + lane];
    float nls1 = lsq[tn1 * 16 + (lane & 15)];

    v8f c0a, c0b;                        // C seeds: every element = l_sq[N]
#pragma unroll
    for (int r = 0; r < 8; ++r) { c0a[r] = ls0; c0b[r] = ls1; }

#pragma unroll
    for (int a = 0; a < MT; ++a) {
      // D = l_sq + (hi*hi + hi*lo + lo*hi), B = -2*L  (f32 accumulate)
      v8f c, d;
      c = __builtin_amdgcn_wmma_f32_16x16x32_f16(false, ahi[a], false, b0h, (short)0, c0a, false, false);
      c = __builtin_amdgcn_wmma_f32_16x16x32_f16(false, ahi[a], false, b0l, (short)0, c,   false, false);
      c = __builtin_amdgcn_wmma_f32_16x16x32_f16(false, alo[a], false, b0h, (short)0, c,   false, false);
      d = __builtin_amdgcn_wmma_f32_16x16x32_f16(false, ahi[a], false, b1h, (short)0, c0b, false, false);
      d = __builtin_amdgcn_wmma_f32_16x16x32_f16(false, ahi[a], false, b1l, (short)0, d,   false, false);
      d = __builtin_amdgcn_wmma_f32_16x16x32_f16(false, alo[a], false, b1h, (short)0, d,   false, false);
#pragma unroll
      for (int r = 0; r < 8; ++r)
        acc[a][r] = fmin3_raw(acc[a][r], c[r], d[r]);
    }

    b0h = n0h; b0l = n0l; ls0 = nls0;
    b1h = n1h; b1l = n1l; ls1 = nls1;
    t0 += 128;
  }

  // C layout: VGPR r -> M=r (lanes 0-15), M=8+r (lanes 16-31); N = lane&15.
  // Min across the 16 lanes of each half (xor masks < 16 stay in-half).
#pragma unroll
  for (int a = 0; a < MT; ++a)
#pragma unroll
    for (int r = 0; r < 8; ++r) {
      float v = acc[a][r];
#pragma unroll
      for (int m = 1; m <= 8; m <<= 1) v = fminf(v, __shfl_xor(v, m, 32));
      acc[a][r] = v;
    }

  if ((lane & 15) == 0) {
    const int h8 = (lane >> 4) * 8;
#pragma unroll
    for (int a = 0; a < MT; ++a)
#pragma unroll
      for (int r = 0; r < 8; ++r) {
        int row = mblk * (MT * 16) + a * 16 + h8 + r;
        atomicMin(rowmin + row, mapf(acc[a][r]));
      }
  }
}

__global__ void k_fin1(const float* __restrict__ usq, const unsigned* __restrict__ rowmin,
                       float* __restrict__ dd, unsigned* __restrict__ mm) {
  int i = blockIdx.x * blockDim.x + threadIdx.x;
  if (i >= N_U) return;
  float us  = usq[i];
  float m   = us + unmapf(rowmin[i]);            // ||u||^2 + min(l_sq - 2 u.l)
  float dv  = sqrtf(fmaxf(m, 0.0f));
  float den = -0.5f * us - 0.5f * (float)NZ * LOG_2PI;
  float v   = expf(den) * (dv + 1e-18f);         // exp(dens + log(div+eps))
  dd[i] = v;
  atomicMin(mm + 0, __float_as_uint(v));         // v >= 0: raw bits ordered
  atomicMax(mm + 1, __float_as_uint(v));
}

__global__ void k_fin2(const float* __restrict__ dd, const unsigned* __restrict__ mm,
                       float* __restrict__ out) {
  int i = blockIdx.x * blockDim.x + threadIdx.x;
  if (i >= N_U) return;
  float mn = __uint_as_float(mm[0]);
  float mx = __uint_as_float(mm[1]);
  // ref: dd -= min(dd); dd /= (max(dd) + eps)  [max taken after the shift]
  out[i] = (dd[i] - mn) / ((mx - mn) + 1e-18f);
}

extern "C" void kernel_launch(void* const* d_in, const int* in_sizes, int n_in,
                              void* d_out, int out_size, void* d_ws, size_t ws_size,
                              hipStream_t stream) {
  (void)in_sizes; (void)n_in; (void)out_size; (void)ws_size;
  const float* Uz = (const float*)d_in[1];   // [4096, 32] f32
  const float* Lz = (const float*)d_in[2];   // [65536, 32] f32  (d_in[0]=pred unused)
  float* out = (float*)d_out;

  // Workspace carve-up (~8.8 MB), 256B-aligned regions.
  char* ws = (char*)d_ws;
  size_t o = 0;
  v16h* LhiP = (v16h*)(ws + o); o += (size_t)LTI * 32 * 32;   // 4 MB
  v16h* LloP = (v16h*)(ws + o); o += (size_t)LTI * 32 * 32;   // 4 MB
  v16h* UhiP = (v16h*)(ws + o); o += (size_t)UT  * 32 * 32;   // 256 KB
  v16h* UloP = (v16h*)(ws + o); o += (size_t)UT  * 32 * 32;   // 256 KB
  float* lsq = (float*)(ws + o); o += (size_t)N_L * 4;        // 256 KB
  float* usq = (float*)(ws + o); o += (size_t)N_U * 4;        // 16 KB
  unsigned* rowmin = (unsigned*)(ws + o); o += (size_t)N_U * 4;
  float* dd  = (float*)(ws + o); o += (size_t)N_U * 4;
  unsigned* mm = (unsigned*)(ws + o); o += 256;

  k_init<<<(N_U + 255) / 256, 256, 0, stream>>>(rowmin, mm);
  k_pack<<<LTI / 8, 256, 0, stream>>>(Lz, LhiP, LloP, lsq, LTI, -2.0f);
  k_pack<<<UT  / 8, 256, 0, stream>>>(Uz, UhiP, UloP, usq, UT,  1.0f);
  dim3 grid(N_U / (MT * 16), 8);               // 64 M-blocks x 8 N-splits
  k_wmma_min<<<grid, 256, 0, stream>>>(UhiP, UloP, LhiP, LloP, lsq, rowmin);
  k_fin1<<<(N_U + 255) / 256, 256, 0, stream>>>(usq, rowmin, dd, mm);
  k_fin2<<<(N_U + 255) / 256, 256, 0, stream>>>(dd, mm, out);
}
